// AdaptiveHyperbolicProjector_27917287424821
// MI455X (gfx1250) — matrix operand order repaired and checked by reading, not compiled, not run on hardware
//
#include <hip/hip_runtime.h>
#include <hip/hip_bf16.h>

#define EPSF      1e-07f
#define MIN_NORMF 1e-15f
#define MAX_NORMF 1000000.0f
#define NN        20000
#define D_IN      512
#define D_HYP     256
#define EDGES     320000
#define VIEWS     7
#define OUT_COLS  2305      // 513 + 7*256
#define KPAD1     544       // 513 padded to multiple of 32

typedef __attribute__((ext_vector_type(16))) _Float16 v16h;
typedef __attribute__((ext_vector_type(8)))  _Float16 v8h;
typedef __attribute__((ext_vector_type(8)))  float    v8f;
typedef __attribute__((ext_vector_type(4)))  unsigned int u32x4;
typedef __attribute__((ext_vector_type(8)))  int      i32x8;
typedef __attribute__((ext_vector_type(4)))  int      i32x4;

// ---------- helpers ----------
__device__ __forceinline__ float wsum(float v) {
    #pragma unroll
    for (int o = 16; o > 0; o >>= 1) v += __shfl_xor(v, o, 32);
    return v;
}
__device__ __forceinline__ float arcoshf(float x) {
    x = fmaxf(x, 1.0f + EPSF);
    return logf(x + sqrtf(x * x - 1.0f));
}
__device__ __forceinline__ void atomAddF(float* p, float v) {
#if defined(__HIP_PLATFORM_AMD__) || defined(__AMDGCN__)
    unsafeAtomicAdd(p, v);     // hardware global_atomic_add_f32
#else
    atomicAdd(p, v);
#endif
}
__device__ __forceinline__ v16h join16(v8h lo, v8h hi) {
    return __builtin_shufflevector(lo, hi, 0,1,2,3,4,5,6,7,8,9,10,11,12,13,14,15);
}

// Issue a TDM 1-D copy of `bytes/2` f16 elements (one row) from global -> LDS.
// D# per CDNA5 ISA: group0 {count=1, lds_addr, global_addr, type=2},
// group1 {data_size=2B, tensor_dim0 = tile_dim0 = nelem, tensor_dim1 = tile_dim1 = 1}.
__device__ __forceinline__ void tdm_copy_to_lds(const void* gsrc, unsigned ldsOff, int nelem) {
    unsigned long long ga = (unsigned long long)(uintptr_t)gsrc;
    u32x4 g0;
    g0.x = 1u;                                        // count = 1 descriptor
    g0.y = ldsOff;                                    // LDS byte address
    g0.z = (unsigned)(ga & 0xFFFFFFFFu);              // global_addr[31:0]
    g0.w = (unsigned)((ga >> 32) & 0x1FFFFFFu) | (2u << 30);  // addr[56:32] | type=2
    i32x8 g1;
    g1[0] = (1 << 16);                                // data_size = 1 -> 2 bytes
    g1[1] = (nelem & 0xFFFF) << 16;                   // tensor_dim0[15:0]
    g1[2] = ((nelem >> 16) & 0xFFFF) | (1 << 16);     // tensor_dim0[31:16] | tensor_dim1=1
    g1[3] = (nelem & 0xFFFF) << 16;                   // tile_dim0 = nelem (fits 16b)
    g1[4] = 1;                                        // tile_dim1 = 1
    g1[5] = nelem;                                    // tensor_dim0_stride (unused, 1 row)
    g1[6] = 0;
    g1[7] = 0;
    i32x4 gz = {};
#if defined(__clang_major__) && (__clang_major__ >= 23)
    i32x8 gz8 = {};
    __builtin_amdgcn_tensor_load_to_lds(g0, g1, gz, gz, gz8, 0);
#else
    __builtin_amdgcn_tensor_load_to_lds(g0, g1, gz, gz, 0);
#endif
}

// ---------- weight packing: W[256][Ksrc] -> P[Kpad/16][256][16] f16 ----------
__global__ __launch_bounds__(256) void k_pack_w(const float* __restrict__ W,
                                                _Float16* __restrict__ P,
                                                int Ksrc, int Kpad) {
    int total = (Kpad >> 4) * 256 * 16;
    for (int idx = blockIdx.x * blockDim.x + threadIdx.x; idx < total;
         idx += gridDim.x * blockDim.x) {
        int j = idx & 15;
        int n = (idx >> 4) & 255;
        int g = idx >> 12;
        int k = g * 16 + j;
        float v = (k < Ksrc) ? W[(size_t)n * Ksrc + k] : 0.0f;
        P[idx] = (_Float16)v;
    }
}

// ---------- zero fill ----------
__global__ __launch_bounds__(256) void k_zero(float* __restrict__ p, int n) {
    for (int i = blockIdx.x * blockDim.x + threadIdx.x; i < n;
         i += gridDim.x * blockDim.x) p[i] = 0.0f;
}

// ---------- WMMA GEMM ----------
// Block = 128 threads (4 waves). Block computes 32 rows x 256 cols? No:
// grid.x = M/32 (m-supertile, 32 rows), grid.y = 2 (each block covers 128 cols,
// one 32-col supertile per wave). Each wave: 32x32 tile = 4 WMMA accumulators.
// A panel (32 x K f16, contiguous since lda==K) staged into LDS via TDM by wave 0;
// A fragments via ds_load_b128, B fragments via global b128 from fragment-major pack.
__global__ __launch_bounds__(128) void k_gemm_f16(
    const _Float16* __restrict__ A, int lda,       // lda == K
    const _Float16* __restrict__ Bp,               // [K/16][256][16]
    const float* __restrict__ bias,                // 256 or null
    float* __restrict__ C, int ldc, int colOff,
    int K) {
    __shared__ _Float16 smem[32 * KPAD1];          // up to 34.8 KB
    const int wave = threadIdx.x >> 5;
    const int lane = threadIdx.x & 31;
    const int kh = lane >> 4;
    const int ml = lane & 15;
    const int mt = blockIdx.x;                     // 32-row supertile
    const int n0 = (blockIdx.y * 4 + wave) * 32;   // wave's 32-col supertile

    if (wave == 0) {
        tdm_copy_to_lds(A + (size_t)mt * 32 * lda,
                        (unsigned)(uintptr_t)&smem[0], 32 * lda);
        __builtin_amdgcn_s_wait_tensorcnt(0);
    }
    __syncthreads();

    v8f acc00 = {}, acc01 = {}, acc10 = {}, acc11 = {};
    const _Float16* As = smem;
    const int nkb = K >> 5;
    for (int kb = 0; kb < nkb; ++kb) {
        const int k0 = kb << 5;
        // A fragments from LDS (16-bit A 16x32 layout)
        const int ab0 = ml * K + k0 + kh * 16;
        const int ab1 = (ml + 16) * K + k0 + kh * 16;
        v16h a0 = join16(*(const v8h*)(As + ab0), *(const v8h*)(As + ab0 + 8));
        v16h a1 = join16(*(const v8h*)(As + ab1), *(const v8h*)(As + ab1 + 8));
        // B fragments from global pack: group g covers 16 K-rows of one column
        const int g = kb * 2 + kh;
        const _Float16* bp0 = Bp + (((size_t)g * 256) + (n0 + ml)) * 16;
        const _Float16* bp1 = Bp + (((size_t)g * 256) + (n0 + 16 + ml)) * 16;
        v16h b0 = *(const v16h*)bp0;
        v16h b1 = *(const v16h*)bp1;
        __builtin_prefetch((const void*)(bp0 + 512 * 16), 0, 3);  // next k-group
        acc00 = __builtin_amdgcn_wmma_f32_16x16x32_f16(false, a0, false, b0, (short)0, acc00, false, false);
        acc01 = __builtin_amdgcn_wmma_f32_16x16x32_f16(false, a0, false, b1, (short)0, acc01, false, false);
        acc10 = __builtin_amdgcn_wmma_f32_16x16x32_f16(false, a1, false, b0, (short)0, acc10, false, false);
        acc11 = __builtin_amdgcn_wmma_f32_16x16x32_f16(false, a1, false, b1, (short)0, acc11, false, false);
    }
    const int col0 = n0 + ml, col1 = n0 + 16 + ml;
    const float bv0 = bias ? bias[col0] : 0.0f;
    const float bv1 = bias ? bias[col1] : 0.0f;
    const int r0 = mt * 32 + kh * 8;               // C layout: VGPR q -> M = q (+8 upper)
    const int r1 = r0 + 16;
    float* C0 = C + (size_t)r0 * ldc + colOff;
    float* C1 = C + (size_t)r1 * ldc + colOff;
    #pragma unroll
    for (int q = 0; q < 8; ++q) {
        C0[(size_t)q * ldc + col0] = acc00[q] + bv0;
        C0[(size_t)q * ldc + col1] = acc01[q] + bv1;
        C1[(size_t)q * ldc + col0] = acc10[q] + bv0;
        C1[(size_t)q * ldc + col1] = acc11[q] + bv1;
    }
}

// ---------- node embed: f1 -> out[:,0:513]; u = logmap0(proj(expmap0(x_tan,c0)),c_in) -> U16 ----------
__global__ __launch_bounds__(256) void k_node_embed(
    const float* __restrict__ emb, const float* __restrict__ tscale,
    const float* __restrict__ curv,
    float* __restrict__ out, _Float16* __restrict__ U16) {
    const int row = blockIdx.x * 8 + (threadIdx.x >> 5);
    if (row >= NN) return;
    const int lane = threadIdx.x & 31;
    const float ts = tscale[0];
    const float K0 = 1.0f / curv[0], sK0 = sqrtf(K0);
    const float K1 = 1.0f / curv[1], sK1 = sqrtf(K1);
    const float* er = emb + (size_t)row * D_IN;
    float ev[17];
    float ss = 0.0f;
    #pragma unroll
    for (int i = 0; i < 17; ++i) {
        int k = lane + 32 * i;
        float e = (k >= 1 && k < 513) ? er[k - 1] : 0.0f;
        ev[i] = e;
        float x = ts * e;
        ss += x * x;
    }
    ss = wsum(ss);
    float xn = fmaxf(sqrtf(ss), MIN_NORMF);
    float th = xn / sK0;
    float sc = sK0 * sinhf(th) / xn;
    float ys = 0.0f;
    #pragma unroll
    for (int i = 0; i < 17; ++i) { float y = sc * ts * ev[i]; ev[i] = y; ys += y * y; }
    ys = wsum(ys);
    float x0 = sqrtf(fmaxf(K0 + ys, EPSF));
    float yn = fmaxf(sqrtf(ys), MIN_NORMF);
    float f  = sK1 * arcoshf(x0 / sK1) / yn;
    float* orow = out + (size_t)row * OUT_COLS;
    _Float16* urow = U16 + (size_t)row * KPAD1;
    #pragma unroll
    for (int i = 0; i < 17; ++i) {
        int k = lane + 32 * i;
        if (k < 513) {
            orow[k] = (k == 0) ? 0.0f : er[k - 1];
            urow[k] = (_Float16)((k == 0) ? 0.0f : (f * ev[i]));
        }
    }
    for (int k = 513 + lane; k < KPAD1; k += 32) urow[k] = (_Float16)0.0f;
}

// ---------- chain1: mv -> h=proj(expmap0) -> mobius_add bias -> ht=logmap0 ----------
__global__ __launch_bounds__(256) void k_chain1(
    const float* __restrict__ MV, const float* __restrict__ bg,
    const float* __restrict__ curv, float* __restrict__ HT) {
    const int row = blockIdx.x * 8 + (threadIdx.x >> 5);
    if (row >= NN) return;
    const int lane = threadIdx.x & 31;
    const float K = 1.0f / curv[1], sK = sqrtf(K);
    const float* mr = MV + (size_t)row * D_HYP;
    float h[8];
    float ss = 0.0f;
    #pragma unroll
    for (int i = 0; i < 8; ++i) {
        int k = lane + 32 * i;
        float m = (k >= 1) ? mr[k] : 0.0f;
        h[i] = m; ss += m * m;
    }
    ss = wsum(ss);
    float xn = fmaxf(sqrtf(ss), MIN_NORMF);
    float sc = sK * sinhf(xn / sK) / xn;
    float hs = 0.0f;
    #pragma unroll
    for (int i = 0; i < 8; ++i) { h[i] *= sc; hs += h[i] * h[i]; }
    hs = wsum(hs);
    float h0 = sqrtf(fmaxf(K + hs, EPSF));
    float yn = fmaxf(sqrtf(hs), MIN_NORMF);
    float b[8]; float bs = 0.0f;
    #pragma unroll
    for (int i = 0; i < 8; ++i) {
        int k = lane + 32 * i;
        float v = (k >= 1) ? bg[k] : 0.0f;
        b[i] = v; bs += v * v;
    }
    bs = wsum(bs);
    float bn = fmaxf(sqrtf(bs), MIN_NORMF);
    float scb = sK * sinhf(bn / sK) / bn;
    float hbs = 0.0f;
    #pragma unroll
    for (int i = 0; i < 8; ++i) { b[i] *= scb; hbs += b[i] * b[i]; }
    hbs = wsum(hbs);
    float hb0 = sqrtf(fmaxf(K + hbs, EPSF));
    float ybn = fmaxf(sqrtf(hbs), MIN_NORMF);
    float fu  = sK * arcoshf(hb0 / sK) / ybn;
    float ad = 0.0f;
    #pragma unroll
    for (int i = 0; i < 8; ++i) ad += (h[i] / yn) * (fu * b[i]);
    ad = wsum(ad);
    float alpha = ad / sK;
    float w[8]; float ux = 0.0f;
    #pragma unroll
    for (int i = 0; i < 8; ++i) {
        float yh = h[i] / yn;
        w[i] = fu * b[i] - alpha * (sK - h0) * yh;
        ux  += h[i] * w[i];
    }
    ux = wsum(ux);
    float t = ux / fmaxf(h0, EPSF);
    float md = 0.0f;
    #pragma unroll
    for (int i = 0; i < 8; ++i) {
        int k = lane + 32 * i;
        float v = (k == 0) ? t : w[i];
        w[i] = v;
        md += v * v;
    }
    md = wsum(md) - 2.0f * t * t;
    float normu = fminf(sqrtf(fmaxf(md, EPSF)), MAX_NORMF);
    float th2 = fmaxf(normu / sK, MIN_NORMF);
    float ch = coshf(th2), sh = sinhf(th2) / th2;
    float r[8]; float rs = 0.0f;
    #pragma unroll
    for (int i = 0; i < 8; ++i) {
        int k = lane + 32 * i;
        float hv = (k == 0) ? h0 : h[i];
        float v = ch * hv + sh * w[i];
        r[i] = v;
        if (k >= 1) rs += v * v;
    }
    rs = wsum(rs);
    float r0  = sqrtf(fmaxf(K + rs, EPSF));
    float yn2 = fmaxf(sqrtf(rs), MIN_NORMF);
    float fl  = sK * arcoshf(r0 / sK) / yn2;
    float* ho = HT + (size_t)row * D_HYP;
    #pragma unroll
    for (int i = 0; i < 8; ++i) {
        int k = lane + 32 * i;
        ho[k] = (k == 0) ? 0.0f : (fl * r[i]);
    }
}

// ---------- scatter-add: AGG[rows[e]] += vals[e]*SRC[cols[e]] ----------
__global__ __launch_bounds__(256) void k_spmm(
    const int* __restrict__ rows, const int* __restrict__ cols,
    const float* __restrict__ vals,   // may be null
    const float* __restrict__ SRC, float* __restrict__ AGG, int E) {
    const int lane = threadIdx.x & 31;
    const int wv = blockIdx.x * 8 + (threadIdx.x >> 5);
    for (int e = wv; e < E; e += gridDim.x * 8) {
        const int r = rows[e], c = cols[e];
        const float v = vals ? vals[e] : 1.0f;
        const float* s = SRC + (size_t)c * D_HYP;
        float* d = AGG + (size_t)r * D_HYP;
        #pragma unroll
        for (int i = 0; i < 8; ++i) {
            int k = lane + 32 * i;
            atomAddF(d + k, v * s[k]);
        }
    }
}

// ---------- mid chain -> f16 ----------
__global__ __launch_bounds__(256) void k_view_mid(
    const float* __restrict__ AGG, const float* __restrict__ curv,
    _Float16* __restrict__ X16) {
    const int row = blockIdx.x * 8 + (threadIdx.x >> 5);
    if (row >= NN) return;
    const int lane = threadIdx.x & 31;
    const float K1 = 1.0f / curv[1], sK1 = sqrtf(K1);
    const float K2 = 1.0f / curv[2], sK2 = sqrtf(K2);
    const float K3 = 1.0f / curv[3], sK3 = sqrtf(K3);
    const float* ar = AGG + (size_t)row * D_HYP;
    float h[8]; float ss = 0.0f;
    #pragma unroll
    for (int i = 0; i < 8; ++i) {
        int k = lane + 32 * i;
        float a = (k >= 1) ? ar[k] : 0.0f;
        h[i] = a; ss += a * a;
    }
    ss = wsum(ss);
    float xn = fmaxf(sqrtf(ss), MIN_NORMF);
    float sc = sK1 * sinhf(xn / sK1) / xn;
    float hs = 0.0f;
    #pragma unroll
    for (int i = 0; i < 8; ++i) { h[i] *= sc; hs += h[i] * h[i]; }
    hs = wsum(hs);
    float h0  = sqrtf(fmaxf(K1 + hs, EPSF));
    float yn1 = fmaxf(sqrtf(hs), MIN_NORMF);
    float fl1 = sK1 * arcoshf(h0 / sK1) / yn1;
    float xs = 0.0f;
    #pragma unroll
    for (int i = 0; i < 8; ++i) { h[i] = fmaxf(fl1 * h[i], 0.0f); xs += h[i] * h[i]; }
    xs = wsum(xs);
    float xn2 = fmaxf(sqrtf(xs), MIN_NORMF);
    float sc2 = sK2 * sinhf(xn2 / sK2) / xn2;
    float hs2 = 0.0f;
    #pragma unroll
    for (int i = 0; i < 8; ++i) { h[i] *= sc2; hs2 += h[i] * h[i]; }
    hs2 = wsum(hs2);
    float h20 = sqrtf(fmaxf(K2 + hs2, EPSF));
    float yn3 = fmaxf(sqrtf(hs2), MIN_NORMF);
    float fe  = sK3 * arcoshf(h20 / sK3) / yn3;
    _Float16* xo = X16 + (size_t)row * D_HYP;
    #pragma unroll
    for (int i = 0; i < 8; ++i) {
        int k = lane + 32 * i;
        xo[k] = (_Float16)((k == 0) ? 0.0f : (fe * h[i]));
    }
}

// ---------- elementwise add + convert ----------
__global__ __launch_bounds__(256) void k_addcvt(const float* __restrict__ A,
                                                const float* __restrict__ B,
                                                _Float16* __restrict__ X, int n) {
    for (int i = blockIdx.x * blockDim.x + threadIdx.x; i < n;
         i += gridDim.x * blockDim.x)
        X[i] = (_Float16)(A[i] + B[i]);
}

// ---------- batchnorm stats: one block per column ----------
__global__ __launch_bounds__(256) void k_bn_stats(const float* __restrict__ Z,
                                                  float* __restrict__ stats) {
    const int col = blockIdx.x;
    float s = 0.0f, s2 = 0.0f;
    for (int r = threadIdx.x; r < NN; r += 256) {
        float z = Z[(size_t)r * D_HYP + col];
        s += z; s2 += z * z;
    }
    __shared__ float sh[512];
    sh[threadIdx.x] = s; sh[256 + threadIdx.x] = s2;
    __syncthreads();
    for (int o = 128; o > 0; o >>= 1) {
        if (threadIdx.x < o) {
            sh[threadIdx.x] += sh[threadIdx.x + o];
            sh[256 + threadIdx.x] += sh[256 + threadIdx.x + o];
        }
        __syncthreads();
    }
    if (threadIdx.x == 0) {
        float mu = sh[0] / (float)NN;
        stats[col] = mu;
        stats[256 + col] = sh[256] / (float)NN - mu * mu;
    }
}

// ---------- bn apply + tanh + convert ----------
__global__ __launch_bounds__(256) void k_bn_tanh(
    const float* __restrict__ Z, const float* __restrict__ stats,
    const float* __restrict__ g, const float* __restrict__ bt,
    _Float16* __restrict__ X, int n) {
    for (int i = blockIdx.x * blockDim.x + threadIdx.x; i < n;
         i += gridDim.x * blockDim.x) {
        int col = i & 255;
        float zn = (Z[i] - stats[col]) * rsqrtf(stats[256 + col] + 1e-5f)
                   * g[col] + bt[col];
        X[i] = (_Float16)tanhf(zn);
    }
}

extern "C" void kernel_launch(void* const* d_in, const int* in_sizes, int n_in,
                              void* d_out, int out_size, void* d_ws, size_t ws_size,
                              hipStream_t stream) {
    const float* emb   = (const float*)d_in[0];
    const int*   adj   = (const int*)  d_in[1];
    const float* avals = (const float*)d_in[2];
    const int*   gin   = (const int*)  d_in[3];
    const float* ts    = (const float*)d_in[4];
    const float* curv  = (const float*)d_in[5];
    const float* Wg    = (const float*)d_in[6];
    const float* bg    = (const float*)d_in[7];
    const float* Wd    = (const float*)d_in[8];
    const float* bd    = (const float*)d_in[9];
    const float* W1    = (const float*)d_in[10];
    const float* b1    = (const float*)d_in[11];
    const float* gam   = (const float*)d_in[12];
    const float* bet   = (const float*)d_in[13];
    const float* W2    = (const float*)d_in[14];
    const float* b2    = (const float*)d_in[15];
    float* out = (float*)d_out;

    char* ws = (char*)d_ws;
    size_t off = 0;
    auto alloc = [&](size_t bytes) { size_t o = off; off += (bytes + 255) & ~(size_t)255; return o; };
    _Float16* U16  = (_Float16*)(ws + alloc((size_t)NN * KPAD1 * 2));
    _Float16* WgP  = (_Float16*)(ws + alloc((size_t)VIEWS * (KPAD1/16) * 256 * 16 * 2));
    _Float16* WdP  = (_Float16*)(ws + alloc((size_t)VIEWS * 16 * 256 * 16 * 2));
    _Float16* W1P  = (_Float16*)(ws + alloc((size_t)VIEWS * 16 * 256 * 16 * 2));
    _Float16* W2P  = (_Float16*)(ws + alloc((size_t)VIEWS * 16 * 256 * 16 * 2));
    float* bufA    = (float*)(ws + alloc((size_t)NN * D_HYP * 4));
    float* bufB    = (float*)(ws + alloc((size_t)NN * D_HYP * 4));
    float* bufC    = (float*)(ws + alloc((size_t)NN * D_HYP * 4));
    _Float16* X16  = (_Float16*)(ws + alloc((size_t)NN * D_HYP * 2));
    float* stats   = (float*)(ws + alloc(512 * 4));

    const int wgpV = (KPAD1/16) * 256 * 16;
    const int sqV  = 16 * 256 * 16;

    for (int v = 0; v < VIEWS; ++v) {
        k_pack_w<<<256, 256, 0, stream>>>(Wg + (size_t)v*256*513, WgP + (size_t)v*wgpV, 513, KPAD1);
        k_pack_w<<<128, 256, 0, stream>>>(Wd + (size_t)v*256*256, WdP + (size_t)v*sqV, 256, 256);
        k_pack_w<<<128, 256, 0, stream>>>(W1 + (size_t)v*256*256, W1P + (size_t)v*sqV, 256, 256);
        k_pack_w<<<128, 256, 0, stream>>>(W2 + (size_t)v*256*256, W2P + (size_t)v*sqV, 256, 256);
    }

    k_node_embed<<<(NN + 7) / 8, 256, 0, stream>>>(emb, ts, curv, out, U16);

    const dim3 gemmGrid(NN / 32, 2);   // 625 m-supertiles x (2 blocks of 4x32 cols)
    const int rowBlocks = (NN + 7) / 8;
    const int nElems    = NN * D_HYP;

    for (int v = 0; v < VIEWS; ++v) {
        const int* rows = adj + (size_t)v * 2 * EDGES;
        const int* cols = rows + EDGES;
        const int* gsrc = gin + (size_t)v * 2 * EDGES;
        const int* gdst = gsrc + EDGES;

        // GEMM1: mv = u_in @ Wg^T  (K = 544)
        k_gemm_f16<<<gemmGrid, 128, 0, stream>>>(U16, KPAD1, WgP + (size_t)v*wgpV,
                                                 nullptr, bufA, D_HYP, 0, KPAD1);
        k_chain1<<<rowBlocks, 256, 0, stream>>>(bufA, bg + (size_t)v*256, curv, bufB);
        k_zero<<<4096, 256, 0, stream>>>(bufC, nElems);
        k_spmm<<<8192, 256, 0, stream>>>(rows, cols, avals + (size_t)v*EDGES, bufB, bufC, EDGES);
        k_view_mid<<<rowBlocks, 256, 0, stream>>>(bufC, curv, X16);
        // GEMM2: e = et @ Wd^T + bd
        k_gemm_f16<<<gemmGrid, 128, 0, stream>>>(X16, D_HYP, WdP + (size_t)v*sqV,
                                                 bd + (size_t)v*256, bufA, D_HYP, 0, D_HYP);
        k_zero<<<4096, 256, 0, stream>>>(bufC, nElems);
        k_spmm<<<8192, 256, 0, stream>>>(gdst, gsrc, nullptr, bufA, bufC, EDGES);
        k_addcvt<<<4096, 256, 0, stream>>>(bufA, bufC, X16, nElems);
        // GEMM3: z = z_in @ W1^T + b1
        k_gemm_f16<<<gemmGrid, 128, 0, stream>>>(X16, D_HYP, W1P + (size_t)v*sqV,
                                                 b1 + (size_t)v*256, bufB, D_HYP, 0, D_HYP);
        k_bn_stats<<<256, 256, 0, stream>>>(bufB, stats);
        k_bn_tanh<<<4096, 256, 0, stream>>>(bufB, stats, gam + (size_t)v*256,
                                            bet + (size_t)v*256, X16, nElems);
        // GEMM4: out_v = t @ W2^T + b2 -> out[:, 513+v*256 : +256]
        k_gemm_f16<<<gemmGrid, 128, 0, stream>>>(X16, D_HYP, W2P + (size_t)v*sqV,
                                                 b2 + (size_t)v*256, out, OUT_COLS,
                                                 513 + v * 256, D_HYP);
    }
}
